// Building_block_7069516169462
// MI455X (gfx1250) — matrix operand order repaired and checked by reading, not compiled, not run on hardware
//
#include <hip/hip_runtime.h>

typedef __attribute__((ext_vector_type(16))) _Float16     v16h;
typedef __attribute__((ext_vector_type(8)))  float        v8f;
typedef __attribute__((ext_vector_type(8)))  unsigned int v8u;

// Problem constants (match reference)
#define BATCH 2
#define NPTS  40960
#define KNB   16
#define HCH   32
#define DOUT  64
#define NW    4      // waves per block
#define PTS   16     // points per wave
#define EPS   1e-5f

__device__ __forceinline__ v8f wmma_f16(v16h a, v16h b, v8f c) {
  return __builtin_amdgcn_wmma_f32_16x16x32_f16(false, a, false, b, (short)0, c,
                                                false, false);
}

// A-matrix 16x32 f16 fragment (ISA 7.12.2 layout): lane m = L&15,
// lanes 0-15: VGPR0..3 -> K=0..7, VGPR4..7 -> K=16..23 ; lanes 16-31: +8.
__device__ __forceinline__ v16h load_a_frag(const _Float16* w, int ldw,
                                            int mbase, int kbase, int lane) {
  const int m  = mbase + (lane & 15);
  const int kb = (lane & 16) ? 8 : 0;
  v8u u;
#pragma unroll
  for (int j = 0; j < 8; ++j) {
    int k = kbase + kb + ((j < 4) ? 2 * j : 16 + 2 * (j - 4));
    u[j] = *(const unsigned int*)(w + m * ldw + k);   // packed (k,k+1)
  }
  return __builtin_bit_cast(v16h, u);
}

// B-matrix 32x16 f16 fragment: col = L&15; lanes 0-15: VGPR0..3 -> K=0..7,
// VGPR4..7 -> K=16..23 ; lanes 16-31: +8 (same order as A).
// Source staged column-major: f[col*ldc + channel].
__device__ __forceinline__ v16h load_b_frag(const _Float16* f, int ldc,
                                            int cbase, int lane) {
  const int ncol = lane & 15;
  const int kb   = (lane & 16) ? 8 : 0;
  v8u u;
#pragma unroll
  for (int j = 0; j < 8; ++j) {
    int c = cbase + kb + ((j < 4) ? 2 * j : 16 + 2 * (j - 4));
    u[j] = *(const unsigned int*)(f + ncol * ldc + c);  // packed (c,c+1)
  }
  return __builtin_bit_cast(v16h, u);
}

#define LDS_FENCE() asm volatile("s_wait_dscnt 0" ::: "memory")

// ---------------------------------------------------------------------------
// Stage 1: rel-pos-enc -> mlp1(+BN fold)+relu -> concat(gathered feature)
//          -> fc1(WMMA)+relu -> softmax_k -> agg -> ap1_mlp(+BN fold)+relu
//          -> f_agg[b][n][32] to workspace
// ---------------------------------------------------------------------------
__global__ __launch_bounds__(128) void stage1_kernel(
    const float* __restrict__ xyz, const float* __restrict__ feature,
    const float* __restrict__ m1w, const float* __restrict__ m1b,
    const float* __restrict__ m1g, const float* __restrict__ m1beta,
    const float* __restrict__ m1m, const float* __restrict__ m1v,
    const float* __restrict__ fcw, const float* __restrict__ fcb,
    const float* __restrict__ apw, const float* __restrict__ apb,
    const float* __restrict__ apg, const float* __restrict__ apbeta,
    const float* __restrict__ apm, const float* __restrict__ apv,
    const int* __restrict__ nidx, float* __restrict__ fagg) {
  __shared__ _Float16 sW1[DOUT * DOUT];          // fc1 weights (f16)
  __shared__ float    sFCB[DOUT];
  __shared__ float    sW1f[HCH * 10];            // BN-folded mlp1
  __shared__ float    sB1f[HCH];
  __shared__ float    sWP[HCH * DOUT];           // BN-folded ap1_mlp
  __shared__ float    sBP[HCH];
  __shared__ _Float16 sFcat[NW][KNB * DOUT];     // [k][c] column-major
  __shared__ float    sScore[NW][DOUT * KNB];    // [row][k]
  __shared__ float    sAgg[NW][DOUT];

  const int tid = threadIdx.x;
  for (int i = tid; i < DOUT * DOUT; i += 128) sW1[i] = (_Float16)fcw[i];
  if (tid < DOUT) sFCB[tid] = fcb[tid];
  if (tid < HCH) {
    float s = m1g[tid] * rsqrtf(m1v[tid] + EPS);
    sB1f[tid] = (m1b[tid] - m1m[tid]) * s + m1beta[tid];
    for (int j = 0; j < 10; ++j) sW1f[tid * 10 + j] = m1w[tid * 10 + j] * s;
    float s2 = apg[tid] * rsqrtf(apv[tid] + EPS);
    sBP[tid] = (apb[tid] - apm[tid]) * s2 + apbeta[tid];
    for (int j = 0; j < DOUT; ++j) sWP[tid * DOUT + j] = apw[tid * DOUT + j] * s2;
  }
  __syncthreads();

  const int w = tid >> 5, lane = tid & 31;
  const int b = blockIdx.y;
  const int k = lane & 15, ch = lane >> 4;
  const int ncol = lane & 15, moff = (lane & 16) ? 8 : 0;
  _Float16* fcat = &sFcat[w][0];
  float*    scr  = &sScore[w][0];
  float*    agv  = &sAgg[w][0];
  const int base = (blockIdx.x * NW + w) * PTS;

  for (int it = 0; it < PTS; ++it) {
    const int    n   = base + it;
    const size_t pnt = (size_t)b * NPTS + n;
    const int    nb  = nidx[pnt * KNB + k];
    const size_t nbp = (size_t)b * NPTS + nb;
    float cx = xyz[pnt * 3 + 0], cy = xyz[pnt * 3 + 1], cz = xyz[pnt * 3 + 2];
    float nx = xyz[nbp * 3 + 0], ny = xyz[nbp * 3 + 1], nz = xyz[nbp * 3 + 2];
    float rx = cx - nx, ry = cy - ny, rz = cz - nz;
    float enc[10] = {sqrtf(rx * rx + ry * ry + rz * rz),
                     rx, ry, rz, cx, cy, cz, nx, ny, nz};
    const float* fb = feature + (size_t)b * HCH * NPTS;
#pragma unroll
    for (int c2 = 0; c2 < 16; ++c2) {
      int c = ch * 16 + c2;
      fcat[k * DOUT + c] = (_Float16)fb[(size_t)c * NPTS + nb];   // f_nb
      float acc = sB1f[c];
#pragma unroll
      for (int j = 0; j < 10; ++j) acc += sW1f[c * 10 + j] * enc[j];
      fcat[k * DOUT + HCH + c] = (_Float16)fmaxf(acc, 0.f);       // f_xyz
    }
    LDS_FENCE();
#pragma unroll
    for (int mt = 0; mt < 4; ++mt) {                              // fc1: 64x64
      v8f acc = {};
#pragma unroll
      for (int kc = 0; kc < 2; ++kc) {
        v16h a  = load_a_frag(sW1, DOUT, mt * 16, kc * 32, lane);
        v16h bb = load_b_frag(fcat, DOUT, kc * 32, lane);
        acc = wmma_f16(a, bb, acc);
      }
#pragma unroll
      for (int v = 0; v < 8; ++v) {
        int r = mt * 16 + moff + v;
        scr[r * KNB + ncol] = fmaxf(acc[v] + sFCB[r], 0.f);
      }
    }
    LDS_FENCE();
#pragma unroll
    for (int h = 0; h < 2; ++h) {                                 // softmax_k + agg
      int r = lane + 32 * h;
      float sv[KNB], mx = -1e30f;
#pragma unroll
      for (int j = 0; j < KNB; ++j) { sv[j] = scr[r * KNB + j]; mx = fmaxf(mx, sv[j]); }
      float den = 0.f, acc = 0.f;
#pragma unroll
      for (int j = 0; j < KNB; ++j) {
        float e = __expf(sv[j] - mx);
        den += e;
        acc += (float)fcat[j * DOUT + r] * e;
      }
      agv[r] = acc / den;
    }
    LDS_FENCE();
    {                                                             // ap1_mlp -> f_agg
      float acc = sBP[lane];
#pragma unroll
      for (int j = 0; j < DOUT; ++j) acc += sWP[lane * DOUT + j] * agv[j];
      fagg[pnt * HCH + lane] = fmaxf(acc, 0.f);
    }
    LDS_FENCE();
  }
}

// ---------------------------------------------------------------------------
// Stage 2: recompute f_xyz -> mlp2(WMMA,+BN fold)+relu, gather f_agg,
//          fc2(WMMA)+relu -> softmax_k -> agg -> ap2_mlp(+BN fold)+relu -> out
// ---------------------------------------------------------------------------
__global__ __launch_bounds__(128) void stage2_kernel(
    const float* __restrict__ xyz,
    const float* __restrict__ m1w, const float* __restrict__ m1b,
    const float* __restrict__ m1g, const float* __restrict__ m1beta,
    const float* __restrict__ m1m, const float* __restrict__ m1v,
    const float* __restrict__ m2w, const float* __restrict__ m2b,
    const float* __restrict__ m2g, const float* __restrict__ m2beta,
    const float* __restrict__ m2m, const float* __restrict__ m2v,
    const float* __restrict__ fcw, const float* __restrict__ fcb,
    const float* __restrict__ apw, const float* __restrict__ apb,
    const float* __restrict__ apg, const float* __restrict__ apbeta,
    const float* __restrict__ apm, const float* __restrict__ apv,
    const int* __restrict__ nidx, const float* __restrict__ fagg,
    float* __restrict__ out) {
  __shared__ _Float16 sW2[DOUT * DOUT];          // fc2 weights (f16)
  __shared__ float    sFCB[DOUT];
  __shared__ float    sW1f[HCH * 10];
  __shared__ float    sB1f[HCH];
  __shared__ _Float16 sM2[HCH * HCH];            // BN-folded mlp2 (f16)
  __shared__ float    sB2f[HCH];
  __shared__ float    sWP[DOUT * DOUT];          // BN-folded ap2_mlp
  __shared__ float    sBP[DOUT];
  __shared__ _Float16 sFx[NW][KNB * HCH];        // [k][c] f_xyz staging
  __shared__ _Float16 sFcat[NW][KNB * DOUT];
  __shared__ float    sScore[NW][DOUT * KNB];
  __shared__ float    sAgg[NW][DOUT];

  const int tid = threadIdx.x;
  for (int i = tid; i < DOUT * DOUT; i += 128) sW2[i] = (_Float16)fcw[i];
  if (tid < HCH) {
    float s1 = m1g[tid] * rsqrtf(m1v[tid] + EPS);
    sB1f[tid] = (m1b[tid] - m1m[tid]) * s1 + m1beta[tid];
    for (int j = 0; j < 10; ++j) sW1f[tid * 10 + j] = m1w[tid * 10 + j] * s1;
    float s2 = m2g[tid] * rsqrtf(m2v[tid] + EPS);
    sB2f[tid] = (m2b[tid] - m2m[tid]) * s2 + m2beta[tid];
    for (int j = 0; j < HCH; ++j)
      sM2[tid * HCH + j] = (_Float16)(m2w[tid * HCH + j] * s2);
  }
  if (tid < DOUT) {
    sFCB[tid] = fcb[tid];
    float s = apg[tid] * rsqrtf(apv[tid] + EPS);
    sBP[tid] = (apb[tid] - apm[tid]) * s + apbeta[tid];
    for (int j = 0; j < DOUT; ++j) sWP[tid * DOUT + j] = apw[tid * DOUT + j] * s;
  }
  __syncthreads();

  const int w = tid >> 5, lane = tid & 31;
  const int b = blockIdx.y;
  const int k = lane & 15, ch = lane >> 4;
  const int ncol = lane & 15, moff = (lane & 16) ? 8 : 0;
  _Float16* fx   = &sFx[w][0];
  _Float16* fcat = &sFcat[w][0];
  float*    scr  = &sScore[w][0];
  float*    agv  = &sAgg[w][0];
  const int base = (blockIdx.x * NW + w) * PTS;

  for (int it = 0; it < PTS; ++it) {
    const int    n   = base + it;
    const size_t pnt = (size_t)b * NPTS + n;
    const int    nb  = nidx[pnt * KNB + k];
    const size_t nbp = (size_t)b * NPTS + nb;
    float cx = xyz[pnt * 3 + 0], cy = xyz[pnt * 3 + 1], cz = xyz[pnt * 3 + 2];
    float nx = xyz[nbp * 3 + 0], ny = xyz[nbp * 3 + 1], nz = xyz[nbp * 3 + 2];
    float rx = cx - nx, ry = cy - ny, rz = cz - nz;
    float enc[10] = {sqrtf(rx * rx + ry * ry + rz * rz),
                     rx, ry, rz, cx, cy, cz, nx, ny, nz};
#pragma unroll
    for (int c2 = 0; c2 < 16; ++c2) {
      int c = ch * 16 + c2;
      fcat[k * DOUT + c] = (_Float16)fagg[nbp * HCH + c];         // f_nb2
      float acc = sB1f[c];                                       // recompute f_xyz
#pragma unroll
      for (int j = 0; j < 10; ++j) acc += sW1f[c * 10 + j] * enc[j];
      fx[k * HCH + c] = (_Float16)fmaxf(acc, 0.f);
    }
    LDS_FENCE();
#pragma unroll
    for (int mt = 0; mt < 2; ++mt) {                              // mlp2: 32x32
      v8f acc = {};
      v16h a  = load_a_frag(sM2, HCH, mt * 16, 0, lane);
      v16h bb = load_b_frag(fx, HCH, 0, lane);
      acc = wmma_f16(a, bb, acc);
#pragma unroll
      for (int v = 0; v < 8; ++v) {
        int r = mt * 16 + moff + v;
        fcat[ncol * DOUT + HCH + r] = (_Float16)fmaxf(acc[v] + sB2f[r], 0.f);
      }
    }
    LDS_FENCE();
#pragma unroll
    for (int mt = 0; mt < 4; ++mt) {                              // fc2: 64x64
      v8f acc = {};
#pragma unroll
      for (int kc = 0; kc < 2; ++kc) {
        v16h a  = load_a_frag(sW2, DOUT, mt * 16, kc * 32, lane);
        v16h bb = load_b_frag(fcat, DOUT, kc * 32, lane);
        acc = wmma_f16(a, bb, acc);
      }
#pragma unroll
      for (int v = 0; v < 8; ++v) {
        int r = mt * 16 + moff + v;
        scr[r * KNB + ncol] = fmaxf(acc[v] + sFCB[r], 0.f);
      }
    }
    LDS_FENCE();
#pragma unroll
    for (int h = 0; h < 2; ++h) {                                 // softmax_k + agg
      int r = lane + 32 * h;
      float sv[KNB], mx = -1e30f;
#pragma unroll
      for (int j = 0; j < KNB; ++j) { sv[j] = scr[r * KNB + j]; mx = fmaxf(mx, sv[j]); }
      float den = 0.f, acc = 0.f;
#pragma unroll
      for (int j = 0; j < KNB; ++j) {
        float e = __expf(sv[j] - mx);
        den += e;
        acc += (float)fcat[j * DOUT + r] * e;
      }
      agv[r] = acc / den;
    }
    LDS_FENCE();
#pragma unroll
    for (int h = 0; h < 2; ++h) {                                 // ap2_mlp -> out
      int c = lane + 32 * h;
      float acc = sBP[c];
#pragma unroll
      for (int j = 0; j < DOUT; ++j) acc += sWP[c * DOUT + j] * agv[j];
      out[((size_t)b * DOUT + c) * NPTS + n] = fmaxf(acc, 0.f);
    }
    LDS_FENCE();
  }
}

extern "C" void kernel_launch(void* const* d_in, const int* in_sizes, int n_in,
                              void* d_out, int out_size, void* d_ws,
                              size_t ws_size, hipStream_t stream) {
  const float* xyz     = (const float*)d_in[0];
  const float* feature = (const float*)d_in[1];
  const float* m1w = (const float*)d_in[2],  *m1b = (const float*)d_in[3];
  const float* m1g = (const float*)d_in[4],  *m1bt = (const float*)d_in[5];
  const float* m1m = (const float*)d_in[6],  *m1v = (const float*)d_in[7];
  const float* f1w = (const float*)d_in[8],  *f1b = (const float*)d_in[9];
  const float* a1w = (const float*)d_in[10], *a1b = (const float*)d_in[11];
  const float* a1g = (const float*)d_in[12], *a1bt = (const float*)d_in[13];
  const float* a1m = (const float*)d_in[14], *a1v = (const float*)d_in[15];
  const float* m2w = (const float*)d_in[16], *m2b = (const float*)d_in[17];
  const float* m2g = (const float*)d_in[18], *m2bt = (const float*)d_in[19];
  const float* m2m = (const float*)d_in[20], *m2v = (const float*)d_in[21];
  const float* f2w = (const float*)d_in[22], *f2b = (const float*)d_in[23];
  const float* a2w = (const float*)d_in[24], *a2b = (const float*)d_in[25];
  const float* a2g = (const float*)d_in[26], *a2bt = (const float*)d_in[27];
  const float* a2m = (const float*)d_in[28], *a2v = (const float*)d_in[29];
  const int*   nidx = (const int*)d_in[30];
  float* out  = (float*)d_out;
  float* fagg = (float*)d_ws;  // B*N*32 floats = 10.5 MB intermediate

  dim3 block(128);
  dim3 grid(NPTS / (NW * PTS), BATCH);  // 640 x 2, exact cover

  stage1_kernel<<<grid, block, 0, stream>>>(
      xyz, feature, m1w, m1b, m1g, m1bt, m1m, m1v,
      f1w, f1b, a1w, a1b, a1g, a1bt, a1m, a1v, nidx, fagg);
  stage2_kernel<<<grid, block, 0, stream>>>(
      xyz, m1w, m1b, m1g, m1bt, m1m, m1v,
      m2w, m2b, m2g, m2bt, m2m, m2v,
      f2w, f2b, a2w, a2b, a2g, a2bt, a2m, a2v, nidx, fagg, out);
}
